// RoPETopKAttention_52364241272961
// MI455X (gfx1250) — matrix operand order repaired and checked by reading, not compiled, not run on hardware
//
#include <hip/hip_runtime.h>
#include <hip/hip_bf16.h>

#define SEQ    2048
#define DMODEL 1024
#define QHEADS 16
#define KVHEADS 4
#define DHEAD  64
#define TOPK   64

typedef __attribute__((ext_vector_type(16))) _Float16 v16h;
typedef __attribute__((ext_vector_type(8)))  _Float16 v8h;
typedef __attribute__((ext_vector_type(8)))  float    v8f;

// ---------- helpers ----------

__device__ inline v16h pack2(const _Float16* p0, const _Float16* p1) {
  v8h a = *(const v8h*)p0;
  v8h b = *(const v8h*)p1;
  v16h r;
#pragma unroll
  for (int e = 0; e < 8; ++e) { r[e] = a[e]; r[e + 8] = b[e]; }
  return r;
}

__device__ inline v8f wmma_f16(v16h a, v16h b, v8f c) {
  return __builtin_amdgcn_wmma_f32_16x16x32_f16(false, a, false, b, (short)0, c,
                                                false, false);
}

__device__ inline float waveSum(float v) {
#pragma unroll
  for (int o = 16; o > 0; o >>= 1) v += __shfl_xor(v, o, 32);
  return v;
}
__device__ inline float waveMax(float v) {
#pragma unroll
  for (int o = 16; o > 0; o >>= 1) v = fmaxf(v, __shfl_xor(v, o, 32));
  return v;
}
__device__ inline int waveSumI(int v) {
#pragma unroll
  for (int o = 16; o > 0; o >>= 1) v += __shfl_xor(v, o, 32);
  return v;
}

__device__ inline float h16bits_to_f32(unsigned short b) {
  return (float)__builtin_bit_cast(_Float16, b);
}
// monotonic 16-bit sort key for f16 (larger key == larger value)
__device__ inline unsigned sortkey(unsigned short h) {
  return (h & 0x8000u) ? (unsigned)((~h) & 0xFFFFu) : (unsigned)(h | 0x8000u);
}
__device__ inline unsigned short unsortkey(unsigned k) {
  return (k & 0x8000u) ? (unsigned short)(k ^ 0x8000u)
                       : (unsigned short)((~k) & 0xFFFFu);
}

// ---------- kernel 0: f32 -> f16 convert ----------

__global__ void f32_to_f16_kernel(const float* __restrict__ src,
                                  _Float16* __restrict__ dst, int n) {
  int i = blockIdx.x * blockDim.x + threadIdx.x;
  if (i < n) dst[i] = (_Float16)src[i];
}

// ---------- kernel 1: fused QKV GEMM (WMMA) ----------
// Q[s,o] = sum_i x[s,i]*Wq[o,i]   (f32 out, RoPE later)
// K[s,o] = sum_i x[s,i]*Wk[o,i]   (f32 out)
// Vt[o,s] = sum_i x[s,i]*Wv[o,i]  (f16, TRANSPOSED for PV GEMM B-frags)
__global__ __launch_bounds__(256) void qkv_gemm_kernel(
    const _Float16* __restrict__ xh, const _Float16* __restrict__ Wq,
    const _Float16* __restrict__ Wk, const _Float16* __restrict__ Wv,
    float* __restrict__ Qf, float* __restrict__ Kf, _Float16* __restrict__ Vt) {
  const int lane = threadIdx.x;        // 0..31
  const int wave = threadIdx.y;        // 0..7
  const int g = lane >> 4, ln = lane & 15;
  const int mbase = blockIdx.y * 16;
  const int nbase = (blockIdx.x * 8 + wave) * 16;  // 0..1535

  const _Float16* W;
  int nloc;
  if (nbase < 1024)      { W = Wq; nloc = nbase; }
  else if (nbase < 1280) { W = Wk; nloc = nbase - 1024; }
  else                   { W = Wv; nloc = nbase - 1280; }

  const _Float16* arow = xh + (size_t)(mbase + ln) * DMODEL;
  const _Float16* brow = W + (size_t)(nloc + ln) * DMODEL;

  v8f acc = {};
  for (int k = 0; k < DMODEL; k += 32) {
    v16h a = pack2(arow + k + 8 * g, arow + k + 16 + 8 * g);
    v16h b = pack2(brow + k + 8 * g, brow + k + 16 + 8 * g);
    acc = wmma_f16(a, b, acc);
  }
#pragma unroll
  for (int r = 0; r < 8; ++r) {
    int m = mbase + r + 8 * g;
    int n = nbase + ln;
    float c = acc[r];
    if (n < 1024)      Qf[(size_t)m * 1024 + n] = c;
    else if (n < 1280) Kf[(size_t)m * 256 + (n - 1024)] = c;
    else               Vt[(size_t)(n - 1280) * SEQ + m] = (_Float16)c;
  }
}

// ---------- kernel 2: RoPE (f32 in -> f16 out) ----------

__global__ void rope_kernel(const float* __restrict__ in,
                            _Float16* __restrict__ out, int nheads) {
  int idx = blockIdx.x * blockDim.x + threadIdx.x;   // pos*heads*32 threads
  int j = idx & 31;
  int h = (idx >> 5) % nheads;
  int pos = idx / (32 * nheads);
  if (pos >= SEQ) return;
  size_t d0 = (size_t)pos * nheads * DHEAD + (size_t)h * DHEAD + j;
  float x1 = in[d0], x2 = in[d0 + 32];
  // inv_freq = 10000^(-j/32) = exp(-j * ln(1e4)/32)
  float freq = expf(-(float)j * (9.210340371976184f / 32.0f));
  float ang = (float)pos * freq;
  float c = cosf(ang), s = sinf(ang);
  out[d0]      = (_Float16)(x1 * c - x2 * s);
  out[d0 + 32] = (_Float16)(x2 * c + x1 * s);
}

// ---------- kernel 3: attention (scores + exact top-64 + softmax + PV) ----------
// grid: (SEQ/16, QHEADS), block: (32, 8)

__global__ __launch_bounds__(256) void attn_kernel(
    const _Float16* __restrict__ Qh, const _Float16* __restrict__ Kh,
    const _Float16* __restrict__ Vt, _Float16* __restrict__ attn) {
  __shared__ _Float16 s_scores[16 * SEQ];          // 64 KB (of 320 KB WGP LDS)
  __shared__ float s_max[16], s_thr[16], s_den[16];

  const int lane = threadIdx.x, wave = threadIdx.y;
  const int g = lane >> 4, ln = lane & 15;
  const int qtile = blockIdx.x, h = blockIdx.y;
  const int kvh = h >> 2;                          // jnp.repeat(k, 4, axis=1)
  const int qbase = qtile * 16;
  const float scale = 0.125f;                      // 1/sqrt(64)

  // Q A-fragments for the two 32-wide k-slabs of DHEAD=64
  const _Float16* qrow = Qh + (size_t)(qbase + ln) * DMODEL + (size_t)h * DHEAD;
  v16h qa0 = pack2(qrow + 8 * g, qrow + 16 + 8 * g);
  v16h qa1 = pack2(qrow + 32 + 8 * g, qrow + 48 + 8 * g);

  // ---- phase 1: scores tile (16 x 2048) -> LDS (f16), causal-masked ----
  for (int kt = wave * 16; kt < wave * 16 + 16; ++kt) {
    int kb = kt * 16;
    const _Float16* krow =
        Kh + (size_t)(kb + ln) * (KVHEADS * DHEAD) + (size_t)kvh * DHEAD;
    v16h b0 = pack2(krow + 8 * g, krow + 16 + 8 * g);
    v16h b1 = pack2(krow + 32 + 8 * g, krow + 48 + 8 * g);
    v8f acc = {};
    acc = wmma_f16(qa0, b0, acc);
    acc = wmma_f16(qa1, b1, acc);
    int key = kb + ln;
#pragma unroll
    for (int r = 0; r < 8; ++r) {
      int m = r + 8 * g;
      float sc = (key <= qbase + m) ? acc[r] * scale : -__builtin_inff();
      s_scores[m * SEQ + key] = (_Float16)sc;
    }
  }
  __syncthreads();

  // ---- phase 2: per-row max, exact 64th-largest (bitwise radix select), denom ----
  const unsigned short* su = (const unsigned short*)s_scores;
  for (int i = 0; i < 2; ++i) {
    int row = wave * 2 + i;
    const unsigned short* rp = su + row * SEQ;
    float mx = -__builtin_inff();
    for (int k = lane; k < SEQ; k += 32) mx = fmaxf(mx, h16bits_to_f32(rp[k]));
    mx = waveMax(mx);

    unsigned cand = 0;
    int remaining = TOPK;
#pragma unroll 1
    for (int bit = 15; bit >= 0; --bit) {
      unsigned test = cand | (1u << bit);
      int cnt = 0;
      for (int k = lane; k < SEQ; k += 32)
        cnt += ((sortkey(rp[k]) >> bit) == (test >> bit)) ? 1 : 0;
      cnt = waveSumI(cnt);
      if (cnt >= remaining) cand = test;
      else remaining -= cnt;
    }
    float thr = h16bits_to_f32(unsortkey(cand));

    float den = 0.f;
    for (int k = lane; k < SEQ; k += 32) {
      float v = h16bits_to_f32(rp[k]);
      if (v >= thr) den += expf(v - mx);
    }
    den = waveSum(den);
    if (lane == 0) { s_max[row] = mx; s_thr[row] = thr; s_den[row] = den; }
  }
  __syncthreads();

  // ---- phase 3: out(16x64) = P(16x2048) @ V(2048x64), keys split over waves ----
  const float tch = s_thr[ln];
  const float mch = s_max[ln];
  v8f acc[4] = {v8f{}, v8f{}, v8f{}, v8f{}};
  const int kstart = wave * 256;
  for (int slab = 0; slab < 8; ++slab) {
    int kb = kstart + slab * 32;
    v16h pa;
#pragma unroll
    for (int e = 0; e < 16; ++e) {
      int kk = (e < 8) ? (e + 8 * g) : (e + 8 + 8 * g);
      float v = h16bits_to_f32(su[ln * SEQ + kb + kk]);
      float w = (v >= tch) ? expf(v - mch) : 0.f;
      pa[e] = (_Float16)w;
    }
#pragma unroll
    for (int nt = 0; nt < 4; ++nt) {
      const _Float16* vrow =
          Vt + (size_t)(kvh * DHEAD + nt * 16 + ln) * SEQ + kb;
      v16h vb = pack2(vrow + 8 * g, vrow + 16 + 8 * g);
      acc[nt] = wmma_f16(pa, vb, acc[nt]);
    }
  }
  __syncthreads();   // done reading s_scores; reuse it as f32 reduction buffer

  float* s_out = (float*)s_scores;  // 16*64 floats
  for (int i = wave * 32 + lane; i < 16 * 64; i += 256) s_out[i] = 0.f;
  __syncthreads();
#pragma unroll
  for (int nt = 0; nt < 4; ++nt)
#pragma unroll
    for (int r = 0; r < 8; ++r) {
      int m = r + 8 * g, n = nt * 16 + ln;
      atomicAdd(&s_out[m * 64 + n], acc[nt][r]);
    }
  __syncthreads();
  for (int i = wave * 32 + lane; i < 16 * 64; i += 256) {
    int m = i >> 6, n = i & 63;
    float val = s_out[i] / s_den[m];
    attn[(size_t)(qbase + m) * DMODEL + (size_t)h * DHEAD + n] = (_Float16)val;
  }
}

// ---------- kernel 4: output projection (WMMA), f32 out ----------

__global__ __launch_bounds__(256) void out_gemm_kernel(
    const _Float16* __restrict__ A, const _Float16* __restrict__ Wo,
    float* __restrict__ out) {
  const int lane = threadIdx.x, wave = threadIdx.y;
  const int g = lane >> 4, ln = lane & 15;
  const int mbase = blockIdx.y * 16;
  const int nbase = (blockIdx.x * 8 + wave) * 16;
  const _Float16* arow = A + (size_t)(mbase + ln) * DMODEL;
  const _Float16* brow = Wo + (size_t)(nbase + ln) * DMODEL;
  v8f acc = {};
  for (int k = 0; k < DMODEL; k += 32) {
    v16h a = pack2(arow + k + 8 * g, arow + k + 16 + 8 * g);
    v16h b = pack2(brow + k + 8 * g, brow + k + 16 + 8 * g);
    acc = wmma_f16(a, b, acc);
  }
#pragma unroll
  for (int r = 0; r < 8; ++r)
    out[(size_t)(mbase + r + 8 * g) * DMODEL + nbase + ln] = acc[r];
}

// ---------- launcher ----------

extern "C" void kernel_launch(void* const* d_in, const int* in_sizes, int n_in,
                              void* d_out, int out_size, void* d_ws,
                              size_t ws_size, hipStream_t stream) {
  const float* x  = (const float*)d_in[0];
  const float* Wq = (const float*)d_in[1];
  const float* Wk = (const float*)d_in[2];
  const float* Wv = (const float*)d_in[3];
  const float* Wo = (const float*)d_in[4];

  char* ws = (char*)d_ws;
  size_t off = 0;
  auto carve = [&](size_t bytes) {
    char* p = ws + off;
    off += (bytes + 255) & ~(size_t)255;
    return p;
  };
  _Float16* xh  = (_Float16*)carve((size_t)SEQ * DMODEL * 2);          // 4 MB
  _Float16* Wqh = (_Float16*)carve((size_t)DMODEL * DMODEL * 2);       // 2 MB
  _Float16* Wkh = (_Float16*)carve((size_t)256 * DMODEL * 2);          // 0.5 MB
  _Float16* Wvh = (_Float16*)carve((size_t)256 * DMODEL * 2);          // 0.5 MB
  _Float16* Woh = (_Float16*)carve((size_t)DMODEL * DMODEL * 2);       // 2 MB
  float*    Qf  = (float*)carve((size_t)SEQ * DMODEL * 4);             // 8 MB
  float*    Kf  = (float*)carve((size_t)SEQ * 256 * 4);                // 2 MB
  _Float16* Qh  = (_Float16*)carve((size_t)SEQ * DMODEL * 2);          // 4 MB
  _Float16* Kh  = (_Float16*)carve((size_t)SEQ * 256 * 2);             // 1 MB
  _Float16* Vt  = (_Float16*)carve((size_t)256 * SEQ * 2);             // 1 MB
  _Float16* Ah  = (_Float16*)carve((size_t)SEQ * DMODEL * 2);          // 4 MB

  // 0) converts
  f32_to_f16_kernel<<<(SEQ * DMODEL + 255) / 256, 256, 0, stream>>>(x, xh, SEQ * DMODEL);
  f32_to_f16_kernel<<<(DMODEL * DMODEL + 255) / 256, 256, 0, stream>>>(Wq, Wqh, DMODEL * DMODEL);
  f32_to_f16_kernel<<<(256 * DMODEL + 255) / 256, 256, 0, stream>>>(Wk, Wkh, 256 * DMODEL);
  f32_to_f16_kernel<<<(256 * DMODEL + 255) / 256, 256, 0, stream>>>(Wv, Wvh, 256 * DMODEL);
  f32_to_f16_kernel<<<(DMODEL * DMODEL + 255) / 256, 256, 0, stream>>>(Wo, Woh, DMODEL * DMODEL);

  // 1) fused QKV GEMM: N = 1024+256+256 = 1536 -> 96 n-tiles / 8 per block
  qkv_gemm_kernel<<<dim3(12, SEQ / 16), dim3(32, 8), 0, stream>>>(
      xh, Wqh, Wkh, Wvh, Qf, Kf, Vt);

  // 2) RoPE
  rope_kernel<<<(SEQ * QHEADS * 32) / 256, 256, 0, stream>>>(Qf, Qh, QHEADS);
  rope_kernel<<<(SEQ * KVHEADS * 32) / 256, 256, 0, stream>>>(Kf, Kh, KVHEADS);

  // 3) attention
  attn_kernel<<<dim3(SEQ / 16, QHEADS), dim3(32, 8), 0, stream>>>(Qh, Kh, Vt, Ah);

  // 4) output projection
  out_gemm_kernel<<<dim3(8, SEQ / 16), dim3(32, 8), 0, stream>>>(
      Ah, Woh, (float*)d_out);
}
